// MatrixRouting_24927990186495
// MI455X (gfx1250) — matrix operand order repaired and checked by reading, not compiled, not run on hardware
//
#include <hip/hip_runtime.h>
#include <hip/hip_bf16.h>

typedef __attribute__((ext_vector_type(16))) _Float16 v16h;
typedef __attribute__((ext_vector_type(8)))  float    v8f;

// async global->LDS builtin operand types (from hipcc diagnostic):
// param 0: int __attribute__((vector_size(16))) __device__ * (AS1, non-const)
typedef int v4i_vs __attribute__((vector_size(16)));
typedef __attribute__((address_space(1))) v4i_vs* as1_b128_t;
typedef __attribute__((address_space(3))) v4i_vs* as3_b128_t;

#define BKK   1152
#define CDIM  32
#define HDIM  16
#define EPSF  1e-10f
#define HALF_LN_2PI 0.91893853320467274178f

#if defined(__has_builtin)
#if __has_builtin(__builtin_amdgcn_global_load_async_to_lds_b128)
#define HAVE_ASYNC_LDS 1
#endif
#if __has_builtin(__builtin_amdgcn_s_wait_asynccnt)
#define HAVE_WAIT_ASYNC_BUILTIN 1
#endif
#endif

__device__ __forceinline__ void wait_asynccnt0() {
#if defined(HAVE_WAIT_ASYNC_BUILTIN)
    __builtin_amdgcn_s_wait_asynccnt(0);
#else
    asm volatile("s_wait_asynccnt 0" ::: "memory");
#endif
}

// One workgroup per batch element. 1024 threads = 32 waves (wave32).
// R (1152x32 f32) lives in LDS for the whole routing loop (CDNA5: 320KB/WGP).
__global__ __launch_bounds__(1024)
void em_routing_kernel(const float* __restrict__ V,
                       const float* __restrict__ aAct,
                       const float* __restrict__ betaV,
                       const float* __restrict__ betaA,
                       const int*   __restrict__ nRoutePtr,
                       float* __restrict__ outMu,
                       float* __restrict__ outA,
                       float* __restrict__ outSumR)
{
    __shared__ float R_lds[BKK * CDIM];      // 147456 B responsibilities
    __shared__ float a_lds[BKK];             // input activations
    __shared__ float mu_s [CDIM * HDIM];     // per-channel mean
    __shared__ float ni2s_s[CDIM * HDIM];    // -1/(2*sigma^2)
    __shared__ float nlc_s[CDIM * HDIM];     // -(log(sqrt(sig)+eps) + 0.5*ln2pi)
    __shared__ float aO_s [CDIM];            // output activation a_c
    __shared__ float den_s[CDIM * HDIM];     // [wave][16] softmax denominators

    const int bI   = blockIdx.x;
    const int tid  = threadIdx.x;
    const int wave = tid >> 5;
    const int lane = tid & 31;
    const int nr   = nRoutePtr[0];

    // ---- init R = 1/C; stage a_act into LDS (async DMA path if available) ----
    for (int n = tid; n < BKK * CDIM; n += 1024) R_lds[n] = 1.0f / 32.0f;
#if defined(HAVE_ASYNC_LDS)
    if (tid < BKK / 4) {   // 288 lanes x 16B, waves 0..8 fully active
        __builtin_amdgcn_global_load_async_to_lds_b128(
            (as1_b128_t)(aAct + (size_t)bI * BKK + tid * 4),
            (as3_b128_t)(a_lds + tid * 4),
            0, 0);
    }
    wait_asynccnt0();
#else
    for (int n = tid; n < BKK; n += 1024) a_lds[n] = aAct[(size_t)bI * BKK + n];
#endif
    __syncthreads();

    const float* Vb = V + (size_t)bI * BKK * CDIM * HDIM;

    // Block-structured B (f16, 32x16): B[k,n] = ((k<16) == (n<8)) ? 1 : 0.
    // In table k order (elements 0-7 <-> K=8h+s, elements 8-15 <-> K=16+8h+(s-8),
    // N = lane%16) element j is 1 iff (j<8)==(N<8): D columns 0..7 = rowsum over K 0..15
    // (channel cc), columns 8..15 = rowsum over K 16..31 (channel cc+1).
    const int nCol  = lane & 15;
    const int kHalf = lane >> 4;                       // 0: K 0..7 & 16..23, 1: K 8..15 & 24..31
    const _Float16 bval =
        (_Float16)((((kHalf == 0) && (nCol < 8)) || ((kHalf == 1) && (nCol >= 8))) ? 1.0f : 0.0f);
    v16h Bblk;
    #pragma unroll
    for (int j = 0; j < 16; ++j) Bblk[j] = (_Float16)((((j < 8)) == (nCol < 8)) ? 1.0f : 0.0f);

    float pdecay = 0.65f;

    for (int iter = 0; iter < nr; ++iter) {
        const float lam = (1.0f - pdecay) * 1.37f;
        pdecay *= 0.65f;

        // ================= M-step: wave w owns channel c = w =================
        const int c = wave;
        float s0 = 0.0f;
        float s1[16], s2[16];
        #pragma unroll
        for (int h = 0; h < 16; ++h) { s1[h] = 0.0f; s2[h] = 0.0f; }

        for (int i = lane; i < BKK; i += 32) {
            float r = R_lds[i * CDIM + c] * a_lds[i];
            const float4* vp = (const float4*)(Vb + ((size_t)i * CDIM + c) * HDIM);
            if (i + 32 < BKK)
                __builtin_prefetch((const void*)(Vb + ((size_t)(i + 32) * CDIM + c) * HDIM), 0, 1);
            float4 x0 = vp[0], x1 = vp[1], x2 = vp[2], x3 = vp[3];
            float vv[16] = { x0.x,x0.y,x0.z,x0.w, x1.x,x1.y,x1.z,x1.w,
                             x2.x,x2.y,x2.z,x2.w, x3.x,x3.y,x3.z,x3.w };
            s0 += r;
            #pragma unroll
            for (int h = 0; h < 16; ++h) {
                float t = vv[h];
                s1[h] = fmaf(r, t, s1[h]);
                s2[h] = fmaf(r * t, t, s2[h]);
            }
        }

        // cross-lane reduce the 33 sufficient statistics (wave32 xor tree)
        #pragma unroll
        for (int m = 16; m >= 1; m >>= 1) {
            s0 += __shfl_xor(s0, m, 32);
            #pragma unroll
            for (int h = 0; h < 16; ++h) {
                s1[h] += __shfl_xor(s1[h], m, 32);
                s2[h] += __shfl_xor(s2[h], m, 32);
            }
        }

        const float sumRc   = s0 + EPSF;
        const float invSumR = 1.0f / sumRc;
        float mu_[16], ni2s_[16], nlc_[16];
        float costAcc = 0.0f;
        const float bVc = betaV[c];
        #pragma unroll
        for (int h = 0; h < 16; ++h) {
            float m_  = s1[h] * invSumR;
            // sum_i R'((V-mu)^2 + eps) = S2 - 2 mu S1 + (mu^2+eps) S0
            float sig = (s2[h] - 2.0f * m_ * s1[h] + (m_ * m_ + EPSF) * s0) * invSumR + EPSF;
            float lsg = logf(sqrtf(sig) + EPSF);
            mu_[h]   = m_;
            ni2s_[h] = -0.5f / sig;
            nlc_[h]  = -(lsg + HALF_LN_2PI);
            costAcc += bVc + lsg;
        }
        const float cost = costAcc * sumRc;
        const float aC   = __builtin_amdgcn_rcpf(1.0f + __expf(-lam * (betaA[c] - cost)));

        if (lane == 0) {
            #pragma unroll
            for (int h = 0; h < 16; ++h) {
                mu_s  [c * 16 + h] = mu_[h];
                ni2s_s[c * 16 + h] = ni2s_[h];
                nlc_s [c * 16 + h] = nlc_[h];
            }
            aO_s[c] = aC;
        }

        if (iter == nr - 1) {
            if (lane == 0) {
                #pragma unroll
                for (int h = 0; h < 16; ++h)
                    outMu[((size_t)bI * CDIM + c) * HDIM + h] = mu_[h];
                outA   [(size_t)bI * CDIM + c] = aC;
                outSumR[(size_t)bI * CDIM + c] = sumRc;
            }
            __syncthreads();
            continue;
        }
        __syncthreads();   // stats visible to all waves

        // ================= E-step: tiles of 16 capsules, 2 channels / WMMA ===
        // A (f16 16x32): lane holds row M=lane%16; lanes 0-15 supply K=0..7 &
        // 16..23 (h 0..7 of cc and cc+1), lanes 16-31 supply K=8..15 & 24..31
        // (h 8..15). Contract with block-B => per-capsule, per-channel row sums.
        const int hBase = kHalf * 8;       // h range this lane computes
        const int iLoc  = lane & 15;       // capsule row within tile
        const int chSel = (nCol >= 8);     // which channel this lane's D holds
        const int wrCh  = (lane >> 3) & 1; // channel written by lanes 0,8,16,24

        for (int tIdx = wave; tIdx < BKK / 16; tIdx += 32) {
            const int i0  = tIdx * 16;
            const int myI = i0 + iLoc;
            v8f dAcc = {};
            for (int cc = 0; cc < CDIM; cc += 2) {
                const float* rowp = Vb + ((size_t)myI * CDIM + cc) * HDIM + hBase;
                const float4* vp0 = (const float4*)(rowp);
                const float4* vp1 = (const float4*)(rowp + HDIM);
                float4 x0 = vp0[0], x1 = vp0[1];      // channel cc
                float4 y0 = vp1[0], y1 = vp1[1];      // channel cc+1
                float vv[16] = { x0.x,x0.y,x0.z,x0.w, x1.x,x1.y,x1.z,x1.w,
                                 y0.x,y0.y,y0.z,y0.w, y1.x,y1.y,y1.z,y1.w };
                v16h Amat;
                #pragma unroll
                for (int j = 0; j < 16; ++j) {
                    const int ch = cc + (j >> 3);
                    const int hh = (j & 7) + hBase;
                    float d   = vv[j] - mu_s[ch * 16 + hh];
                    float vmm = fmaf(d, d, EPSF);
                    float lnp = fmaf(vmm, ni2s_s[ch * 16 + hh], nlc_s[ch * 16 + hh]);
                    Amat[j]   = (_Float16)__expf(lnp);
                }
                v8f cz = {};
                v8f D = __builtin_amdgcn_wmma_f32_16x16x32_f16(
                    false, Amat, false, Bblk, (short)0, cz, false, false);
                const float aSel = aO_s[cc + chSel];
                #pragma unroll
                for (int v = 0; v < 8; ++v) dAcc[v] += aSel * D[v];
                // lanes 0,8,16,24 (columns n=0 and n=8 of each half) commit
                // ap = a_ch * rowsum for capsules i0 + v + 8*kHalf.
                if ((lane & 7) == 0) {
                    const float aW = aO_s[cc + wrCh];
                    #pragma unroll
                    for (int v = 0; v < 8; ++v)
                        R_lds[(i0 + v + 8 * kHalf) * CDIM + (cc + wrCh)] = aW * D[v];
                }
            }
            // merge even/odd channel denominator partials, then publish
            #pragma unroll
            for (int v = 0; v < 8; ++v) dAcc[v] += __shfl_xor(dAcc[v], 8, 32);
            if ((lane & 15) == 0) {
                #pragma unroll
                for (int v = 0; v < 8; ++v)
                    den_s[wave * 16 + v + 8 * kHalf] = dAcc[v];
            }
            // normalize: lane <-> channel, fast reciprocal (v_rcp_f32) + FMA
            #pragma unroll 4
            for (int il = 0; il < 16; ++il) {
                float rdn = __builtin_amdgcn_rcpf(den_s[wave * 16 + il] + EPSF);
                float ap  = R_lds[(i0 + il) * CDIM + lane];
                R_lds[(i0 + il) * CDIM + lane] = fmaf(ap, rdn, EPSF);
            }
        }
        __syncthreads();   // R ready for next M-step
    }
}

extern "C" void kernel_launch(void* const* d_in, const int* in_sizes, int n_in,
                              void* d_out, int out_size, void* d_ws, size_t ws_size,
                              hipStream_t stream) {
    const float* V  = (const float*)d_in[0];
    const float* aA = (const float*)d_in[1];
    const float* bV = (const float*)d_in[2];
    const float* bA = (const float*)d_in[3];
    const int*   nr = (const int*)d_in[4];

    const int b = in_sizes[0] / (BKK * CDIM * HDIM);

    float* out   = (float*)d_out;
    float* outMu = out;                                   // [b, C, 16]
    float* outA  = out + (size_t)b * CDIM * HDIM;         // [b, C, 1]
    float* outS  = outA + (size_t)b * CDIM;               // [b, C]

    em_routing_kernel<<<dim3(b), dim3(1024), 0, stream>>>(
        V, aA, bV, bA, nr, outMu, outA, outS);
}
